// CustomAttention_38543036514925
// MI455X (gfx1250) — compile-verified, matrix operand order checked by reading, not run on hardware
//
#include <hip/hip_runtime.h>
#include <stdint.h>

// ---------------------------------------------------------------------------
// Types (avoid HIP vector structs inside unions; use raw clang ext vectors)
// ---------------------------------------------------------------------------
typedef __attribute__((ext_vector_type(16))) __bf16        v16bf;
typedef __attribute__((ext_vector_type(8)))  float         v8f;
typedef __attribute__((ext_vector_type(4)))  unsigned int  u32x4;
typedef __attribute__((ext_vector_type(2)))  unsigned int  u32x2;
typedef __attribute__((ext_vector_type(4)))  float         f32x4;

static constexpr int B_  = 32;
static constexpr int N_  = 577;           // tokens (1 CLS + 24x24 patches)
static constexpr int C_  = 768;
static constexpr int H_  = 12;
static constexpr int D_  = 64;
static constexpr int M_  = B_ * N_;       // 18464 rows in the token-major GEMMs
static constexpr int NP  = 608;           // tokens padded to 19*32 for key tiles
static constexpr int QT  = 37;            // ceil(577/16) query tiles
static constexpr int KT  = 19;            // 608/32 key tiles
static constexpr int GRID = 24;           // patch grid

// ---------------------------------------------------------------------------
// Helpers
// ---------------------------------------------------------------------------
__device__ __forceinline__ int iabs_(int v) { return v < 0 ? -v : v; }

// fp32 -> bf16 bits, round-to-nearest-even
__device__ __forceinline__ unsigned short f2bf(float x) {
  union { float f; uint32_t u; } v; v.f = x;
  uint32_t r = v.u + 0x7FFFu + ((v.u >> 16) & 1u);
  return (unsigned short)(r >> 16);
}

// WMMA A/B fragment loader. CDNA5 16-bit 16x32 layout: lane holds row (A) /
// col (B) = lane&15; the 16 bf16 per lane are two contiguous 8-element
// K-chunks at K = half*8 and K = 16 + half*8 (half = lane>>4). With data
// stored K-contiguous per row this is exactly two 16-byte loads.
union FragU { v16bf v; u32x4 q[2]; };

__device__ __forceinline__ v16bf load_frag(const unsigned short* row, int kh) {
  FragU f;
  f.q[0] = *reinterpret_cast<const u32x4*>(row + kh * 8);
  f.q[1] = *reinterpret_cast<const u32x4*>(row + 16 + kh * 8);
  return f.v;
}

__device__ __forceinline__ v8f wmma_bf16(v16bf a, v16bf b, v8f c) {
  // D = A(16x32) * B(32x16) + C, f32 accumulate
  return __builtin_amdgcn_wmma_f32_16x16x32_bf16(
      false, a, false, b, (short)0, c, false, false);
}

// ---------------------------------------------------------------------------
// fp32 -> bf16 bulk convert (n % 4 == 0 for all our tensors)
// ---------------------------------------------------------------------------
__global__ void cvt_bf16(const float* __restrict__ in,
                         unsigned short* __restrict__ out, int n) {
  int i = (blockIdx.x * blockDim.x + threadIdx.x) * 4;
  if (i >= n) return;
  f32x4 v = *reinterpret_cast<const f32x4*>(in + i);
  u32x2 p;
  p[0] = (unsigned)f2bf(v[0]) | ((unsigned)f2bf(v[1]) << 16);
  p[1] = (unsigned)f2bf(v[2]) | ((unsigned)f2bf(v[3]) << 16);
  *reinterpret_cast<u32x2*>(out + i) = p;
}

// Zero the token-padding regions of q/k (token-major) and vT (d-major).
__global__ void zero_pads(unsigned short* __restrict__ qb,
                          unsigned short* __restrict__ kb,
                          unsigned short* __restrict__ vtb) {
  const int total = B_ * H_ * (NP - N_) * D_;
  int idx = blockIdx.x * blockDim.x + threadIdx.x;
  if (idx >= total) return;
  int d    = idx & 63;
  int rest = idx >> 6;
  int t    = rest % (NP - N_);
  int bh   = rest / (NP - N_);
  int tok  = N_ + t;
  qb [((size_t)bh * NP + tok) * D_ + d]  = 0;
  kb [((size_t)bh * NP + tok) * D_ + d]  = 0;
  vtb[((size_t)bh * D_ + d) * NP + tok]  = 0;
}

// ---------------------------------------------------------------------------
// QKV GEMM: Y[m,n] = sum_k X[m,k] * Wqkv[n,k] + b[n],  M=18464, K=768, N=2304
// Block tile 128x64, 8 waves (4x2), each wave 32x32 = 2x2 WMMA tiles.
// Epilogue scatters into q [bh,tok,d] (*scale), k [bh,tok,d], vT [bh,d,tok].
// ---------------------------------------------------------------------------
__global__ __launch_bounds__(256)
void qkv_gemm(const unsigned short* __restrict__ xb,
              const unsigned short* __restrict__ wb,
              const float* __restrict__ bias,
              unsigned short* __restrict__ qb,
              unsigned short* __restrict__ kb,
              unsigned short* __restrict__ vtb) {
  const int lane   = threadIdx.x & 31;
  const int wid    = threadIdx.x >> 5;
  const int wm     = wid >> 1, wn = wid & 1;
  const int lanelo = lane & 15, kh = lane >> 4;
  const int m0 = blockIdx.x * 128 + wm * 32;
  const int n0 = blockIdx.y * 64  + wn * 32;

  v8f acc[2][2] = {};
  const int rA0 = min(m0 + lanelo,      M_ - 1);   // clamp: OOB rows read row
  const int rA1 = min(m0 + 16 + lanelo, M_ - 1);   // M-1, discarded at store
  const unsigned short* a0r = xb + (size_t)rA0 * C_;
  const unsigned short* a1r = xb + (size_t)rA1 * C_;
  const unsigned short* b0r = wb + (size_t)(n0 + lanelo) * C_;
  const unsigned short* b1r = wb + (size_t)(n0 + 16 + lanelo) * C_;

#pragma unroll 4
  for (int kk = 0; kk < C_; kk += 32) {
    v16bf a0 = load_frag(a0r + kk, kh);
    v16bf a1 = load_frag(a1r + kk, kh);
    v16bf b0 = load_frag(b0r + kk, kh);
    v16bf b1 = load_frag(b1r + kk, kh);
    acc[0][0] = wmma_bf16(a0, b0, acc[0][0]);
    acc[0][1] = wmma_bf16(a0, b1, acc[0][1]);
    acc[1][0] = wmma_bf16(a1, b0, acc[1][0]);
    acc[1][1] = wmma_bf16(a1, b1, acc[1][1]);
  }

#pragma unroll
  for (int j = 0; j < 2; ++j) {
    const int n     = n0 + j * 16 + lanelo;     // < 2304
    const int which = n / C_;                   // 0=q 1=k 2=v (uniform per blk)
    const int c     = n - which * C_;
    const int h     = c >> 6, d = c & 63;
    const float bv  = bias[n];
#pragma unroll
    for (int i = 0; i < 2; ++i) {
#pragma unroll
      for (int r = 0; r < 8; ++r) {
        const int m = m0 + i * 16 + r + 8 * kh; // C layout: M = r + 8*half
        if (m >= M_) continue;
        const int bi  = m / N_;
        const int tok = m - bi * N_;
        const int bh  = bi * H_ + h;
        float y = acc[i][j][r] + bv;
        if (which == 0)
          qb[((size_t)bh * NP + tok) * D_ + d] = f2bf(y * 0.125f); // q*scale
        else if (which == 1)
          kb[((size_t)bh * NP + tok) * D_ + d] = f2bf(y);
        else
          vtb[((size_t)bh * D_ + d) * NP + tok] = f2bf(y);         // V^T
      }
    }
  }
}

// ---------------------------------------------------------------------------
// Windowed flash attention: one wave per (bh, 16-query tile).
// ---------------------------------------------------------------------------
__device__ __forceinline__ bool mask_ok(int qi, int kj) {
  if (kj >= N_) return false;            // padded keys
  if (qi == 0 || kj == 0) return true;   // CLS row/col
  int pq = qi - 1, pk = kj - 1;
  int qr = pq / GRID, qc = pq - qr * GRID;
  int kr = pk / GRID, kc = pk - kr * GRID;
  return (iabs_(qr - kr) <= 1) && (iabs_(qc - kc) <= 1);
}

__global__ __launch_bounds__(32)
void attn_kernel(const unsigned short* __restrict__ qb,
                 const unsigned short* __restrict__ kb,
                 const unsigned short* __restrict__ vtb,
                 unsigned short* __restrict__ aout) {
  __shared__ unsigned short plds[16 * 32];   // P tile, C-layout -> A-layout
  const int lane   = threadIdx.x & 31;
  const int lanelo = lane & 15, kh = lane >> 4;
  const int qt = blockIdx.x;
  const int bh = blockIdx.y;
  const int bi = bh / H_, h = bh - bi * H_;
  const int q0 = qt * 16;

  // Q fragments (q pre-scaled by D^-0.5), K-dim = 64 -> two 32-wide steps
  const unsigned short* qrow = qb + ((size_t)bh * NP + q0 + lanelo) * D_;
  const v16bf qf0 = load_frag(qrow, kh);
  const v16bf qf1 = load_frag(qrow + 32, kh);

  v8f o[4] = {};                 // 16 x 64 output accumulator (4 n-tiles)
  float rm[8], rs[8];            // per-row running max / sum (row = r + 8*kh)
#pragma unroll
  for (int r = 0; r < 8; ++r) { rm[r] = -1e30f; rs[r] = 0.f; }

  for (int kt = 0; kt < KT; ++kt) {
    const int k0 = kt * 32;
    // Window sparsity: |row delta|<=1 -> |token delta|<=25; CLS handled by
    // kt==0 (key 0) and qt==0 (query 0). Uniform branch -> s_cbranch.
    if (kt > 0 && qt > 0) {
      if (k0 > q0 + 15 + 25 || k0 + 31 < q0 - 25) continue;
    }

    // S = Q K^T : 2 n-tiles x 2 k-steps = 4 WMMAs
    const unsigned short* k0r = kb + ((size_t)bh * NP + k0 + lanelo) * D_;
    const unsigned short* k1r = kb + ((size_t)bh * NP + k0 + 16 + lanelo) * D_;
    v8f s0 = {}, s1 = {};
    s0 = wmma_bf16(qf0, load_frag(k0r,      kh), s0);
    s0 = wmma_bf16(qf1, load_frag(k0r + 32, kh), s0);
    s1 = wmma_bf16(qf0, load_frag(k1r,      kh), s1);
    s1 = wmma_bf16(qf1, load_frag(k1r + 32, kh), s1);

    const int kj0 = k0 + lanelo, kj1 = k0 + 16 + lanelo;
#pragma unroll
    for (int r = 0; r < 8; ++r) {
      const int qi = q0 + r + 8 * kh;
      float a = mask_ok(qi, kj0) ? s0[r] : -1e30f;
      float b = mask_ok(qi, kj1) ? s1[r] : -1e30f;
      // row max across the 16 lanes of this half + both n-tiles
      float mx = fmaxf(a, b);
      mx = fmaxf(mx, __shfl_xor(mx, 1));
      mx = fmaxf(mx, __shfl_xor(mx, 2));
      mx = fmaxf(mx, __shfl_xor(mx, 4));
      mx = fmaxf(mx, __shfl_xor(mx, 8));
      const float mnew = fmaxf(rm[r], mx);
      const float corr = __expf(rm[r] - mnew);
      const float p0 = __expf(a - mnew);
      const float p1 = __expf(b - mnew);
      float ps = p0 + p1;
      ps += __shfl_xor(ps, 1);
      ps += __shfl_xor(ps, 2);
      ps += __shfl_xor(ps, 4);
      ps += __shfl_xor(ps, 8);
      rs[r] = rs[r] * corr + ps;
      rm[r] = mnew;
#pragma unroll
      for (int j = 0; j < 4; ++j) o[j][r] *= corr;   // rescale accumulators
      // park P in LDS (row-major [16][32] bf16) for the A-layout reload
      const int row = r + 8 * kh;
      plds[row * 32 + lanelo]      = f2bf(p0);
      plds[row * 32 + 16 + lanelo] = f2bf(p1);
    }
    asm volatile("s_wait_dscnt 0" ::: "memory");     // P visible across lanes

    // O += P(16x32) * V(32x64): V^T rows are K(=key)-contiguous in global
    const v16bf pf = load_frag(&plds[lanelo * 32], kh);
#pragma unroll
    for (int j = 0; j < 4; ++j) {
      const unsigned short* vrow =
          vtb + ((size_t)bh * D_ + j * 16 + lanelo) * NP + k0;
      o[j] = wmma_bf16(pf, load_frag(vrow, kh), o[j]);
    }
  }

  // normalize + store head-concat output [b, tok, h*64+d] as bf16
#pragma unroll
  for (int r = 0; r < 8; ++r) {
    const int qi = q0 + r + 8 * kh;
    if (qi >= N_) continue;
    const float inv = 1.0f / rs[r];
#pragma unroll
    for (int j = 0; j < 4; ++j) {
      aout[((size_t)bi * N_ + qi) * C_ + h * D_ + j * 16 + lanelo] =
          f2bf(o[j][r] * inv);
    }
  }
}

// ---------------------------------------------------------------------------
// Output projection: out[m,n] = sum_k A[m,k]*Wp[n,k] + b[n], fp32 out
// ---------------------------------------------------------------------------
__global__ __launch_bounds__(256)
void proj_gemm(const unsigned short* __restrict__ ab,
               const unsigned short* __restrict__ wb,
               const float* __restrict__ bias,
               float* __restrict__ out) {
  const int lane   = threadIdx.x & 31;
  const int wid    = threadIdx.x >> 5;
  const int wm     = wid >> 1, wn = wid & 1;
  const int lanelo = lane & 15, kh = lane >> 4;
  const int m0 = blockIdx.x * 128 + wm * 32;
  const int n0 = blockIdx.y * 64  + wn * 32;

  v8f acc[2][2] = {};
  const int rA0 = min(m0 + lanelo,      M_ - 1);
  const int rA1 = min(m0 + 16 + lanelo, M_ - 1);
  const unsigned short* a0r = ab + (size_t)rA0 * C_;
  const unsigned short* a1r = ab + (size_t)rA1 * C_;
  const unsigned short* b0r = wb + (size_t)(n0 + lanelo) * C_;
  const unsigned short* b1r = wb + (size_t)(n0 + 16 + lanelo) * C_;

#pragma unroll 4
  for (int kk = 0; kk < C_; kk += 32) {
    v16bf a0 = load_frag(a0r + kk, kh);
    v16bf a1 = load_frag(a1r + kk, kh);
    v16bf b0 = load_frag(b0r + kk, kh);
    v16bf b1 = load_frag(b1r + kk, kh);
    acc[0][0] = wmma_bf16(a0, b0, acc[0][0]);
    acc[0][1] = wmma_bf16(a0, b1, acc[0][1]);
    acc[1][0] = wmma_bf16(a1, b0, acc[1][0]);
    acc[1][1] = wmma_bf16(a1, b1, acc[1][1]);
  }

#pragma unroll
  for (int j = 0; j < 2; ++j) {
    const int n    = n0 + j * 16 + lanelo;
    const float bv = bias[n];
#pragma unroll
    for (int i = 0; i < 2; ++i) {
#pragma unroll
      for (int r = 0; r < 8; ++r) {
        const int m = m0 + i * 16 + r + 8 * kh;
        if (m < M_) out[(size_t)m * C_ + n] = acc[i][j][r] + bv;
      }
    }
  }
}

// ---------------------------------------------------------------------------
// Launch
// ---------------------------------------------------------------------------
extern "C" void kernel_launch(void* const* d_in, const int* in_sizes, int n_in,
                              void* d_out, int out_size, void* d_ws, size_t ws_size,
                              hipStream_t stream) {
  const float* x      = (const float*)d_in[0];
  const float* qkv_w  = (const float*)d_in[1];
  const float* qkv_b  = (const float*)d_in[2];
  const float* proj_w = (const float*)d_in[3];
  const float* proj_b = (const float*)d_in[4];
  float* out = (float*)d_out;

  // Workspace carve-out (~151 MB), 256B-aligned slices.
  char* ws = (char*)d_ws;
  size_t off = 0;
  auto take = [&](size_t bytes) -> char* {
    char* p = ws + off;
    off += (bytes + 255) & ~(size_t)255;
    return p;
  };
  unsigned short* xb    = (unsigned short*)take((size_t)M_ * C_ * 2);
  unsigned short* wqkv  = (unsigned short*)take((size_t)3 * C_ * C_ * 2);
  unsigned short* wproj = (unsigned short*)take((size_t)C_ * C_ * 2);
  unsigned short* qb    = (unsigned short*)take((size_t)B_ * H_ * NP * D_ * 2);
  unsigned short* kb    = (unsigned short*)take((size_t)B_ * H_ * NP * D_ * 2);
  unsigned short* vtb   = (unsigned short*)take((size_t)B_ * H_ * NP * D_ * 2);
  unsigned short* aout  = (unsigned short*)take((size_t)M_ * C_ * 2);

  const int nx  = M_ * C_;
  const int nw1 = 3 * C_ * C_;
  const int nw2 = C_ * C_;
  cvt_bf16<<<(nx  / 4 + 255) / 256, 256, 0, stream>>>(x,      xb,    nx);
  cvt_bf16<<<(nw1 / 4 + 255) / 256, 256, 0, stream>>>(qkv_w,  wqkv,  nw1);
  cvt_bf16<<<(nw2 / 4 + 255) / 256, 256, 0, stream>>>(proj_w, wproj, nw2);

  const int npad = B_ * H_ * (NP - N_) * D_;
  zero_pads<<<(npad + 255) / 256, 256, 0, stream>>>(qb, kb, vtb);

  dim3 g1((M_ + 127) / 128, (3 * C_) / 64);          // 145 x 36
  qkv_gemm<<<g1, 256, 0, stream>>>(xb, wqkv, qkv_b, qb, kb, vtb);

  dim3 g2(QT, B_ * H_);                              // 37 x 384
  attn_kernel<<<g2, 32, 0, stream>>>(qb, kb, vtb, aout);

  dim3 g3((M_ + 127) / 128, C_ / 64);                // 145 x 12
  proj_gemm<<<g3, 256, 0, stream>>>(aout, wproj, proj_b, out);
}
